// T5Attention_10780367913675
// MI455X (gfx1250) — matrix operand order repaired and checked by reading, not compiled, not run on hardware
//
#include <hip/hip_runtime.h>
#include <math.h>

#define D_MODEL 1024
#define NUM_HEADS 16
#define D_K 64
#define SEQ 2048
#define BATCH 2
#define NUM_BUCKETS 32

typedef __bf16 v16bf __attribute__((ext_vector_type(16)));
typedef float  v8f   __attribute__((ext_vector_type(8)));
typedef unsigned int u32x4 __attribute__((ext_vector_type(4)));

union BFrag {
    v16bf v;
    unsigned short u[16];
    u32x4 q[2];
};

__device__ inline u32x4 ld_g4(const void* p) {
    return *reinterpret_cast<const u32x4*>(p);
}

// native f32 -> bf16 (compiler lowers fptrunc to hw cvt on gfx1250)
__device__ inline unsigned short bf16u(float f) {
    __bf16 h = (__bf16)f;
    return __builtin_bit_cast(unsigned short, h);
}

__device__ inline v8f wmma_bf16(v16bf a, v16bf b, v8f c) {
    return __builtin_amdgcn_wmma_f32_16x16x32_bf16(
        /*neg_a=*/false, a, /*neg_b=*/false, b,
        /*c_mod=*/(short)0, c, /*reuse_a=*/false, /*reuse_b=*/false);
}

__device__ inline int rel_bucket(int d) {   // d = kpos - qpos (mem - ctx)
    int base = (d != 0) ? 16 : 0;
    int a = d < 0 ? -d : d;
    int small = a;
    if (a >= 8) {
        // 8 + trunc(log(a/8)/log(64) * 8), clamped to 15
        int large = 8 + (int)(__logf((float)a * 0.125f) * (8.0f / 4.158883083359672f));
        small = large < 15 ? large : 15;
    }
    return base + small;
}

// ---------------------------------------------------------------------------
// K0a: transpose+convert the four weight matrices: WT[n][k] = bf16(W[k][n])
// ---------------------------------------------------------------------------
__global__ void t5_wtrans(const float* __restrict__ Wq, const float* __restrict__ Wk,
                          const float* __restrict__ Wv, const float* __restrict__ Wo,
                          unsigned short* __restrict__ wtq, unsigned short* __restrict__ wtk,
                          unsigned short* __restrict__ wtv, unsigned short* __restrict__ wto) {
    const float* W;
    unsigned short* WT;
    switch (blockIdx.y) {
        case 0:  W = Wq; WT = wtq; break;
        case 1:  W = Wk; WT = wtk; break;
        case 2:  W = Wv; WT = wtv; break;
        default: W = Wo; WT = wto; break;
    }
    int idx = blockIdx.x * 256 + threadIdx.x;      // k-major for coalesced read
    int k = idx >> 10;
    int n = idx & (D_MODEL - 1);
    WT[(size_t)n * D_MODEL + k] = bf16u(W[(size_t)k * D_MODEL + n]);
}

// ---------------------------------------------------------------------------
// K0b: convert activations f32 -> bf16 (query/key/value), straight layout
// ---------------------------------------------------------------------------
__global__ void t5_xcvt(const float* __restrict__ q_in, const float* __restrict__ k_in,
                        const float* __restrict__ v_in,
                        unsigned short* __restrict__ xq, unsigned short* __restrict__ xk,
                        unsigned short* __restrict__ xv) {
    const float* X = (blockIdx.y == 0) ? q_in : (blockIdx.y == 1) ? k_in : v_in;
    unsigned short* O = (blockIdx.y == 0) ? xq : (blockIdx.y == 1) ? xk : xv;
    size_t idx = (size_t)blockIdx.x * 1024 + threadIdx.x * 4;   // 4 elems/thread
    float4 f = *reinterpret_cast<const float4*>(X + idx);
    unsigned short o0 = bf16u(f.x), o1 = bf16u(f.y), o2 = bf16u(f.z), o3 = bf16u(f.w);
    unsigned int lo = (unsigned int)o0 | ((unsigned int)o1 << 16);
    unsigned int hi = (unsigned int)o2 | ((unsigned int)o3 << 16);
    *reinterpret_cast<unsigned int*>(O + idx)     = lo;
    *reinterpret_cast<unsigned int*>(O + idx + 2) = hi;
}

// ---------------------------------------------------------------------------
// K1: fused QKV projection. One wave per 16x64 output tile (4 WMMA columns),
// K = 1024.  q,k -> [b][h][s][dk] bf16 ; v -> [b][h][dk][s] bf16 (transposed)
// ---------------------------------------------------------------------------
__global__ void t5_qkv(const unsigned short* __restrict__ xq,
                       const unsigned short* __restrict__ xk,
                       const unsigned short* __restrict__ xv,
                       const unsigned short* __restrict__ wtq,
                       const unsigned short* __restrict__ wtk,
                       const unsigned short* __restrict__ wtv,
                       const float* __restrict__ bq, const float* __restrict__ bk,
                       const float* __restrict__ bv,
                       unsigned short* __restrict__ qh, unsigned short* __restrict__ kh,
                       unsigned short* __restrict__ vth) {
    const int lane = threadIdx.x & 31;
    const int m0   = blockIdx.x * 16;
    const int n0   = blockIdx.y * 64;
    const int mat  = blockIdx.z;

    const unsigned short* X  = (mat == 0) ? xq  : (mat == 1) ? xk  : xv;
    const unsigned short* WT = (mat == 0) ? wtq : (mat == 1) ? wtk : wtv;
    const float* bias        = (mat == 0) ? bq  : (mat == 1) ? bk  : bv;

    const int row = lane & 15;
    const int hi  = lane >> 4;
    const unsigned short* xrow = X + (size_t)(m0 + row) * D_MODEL;
    const unsigned short* wrow[4];
    #pragma unroll
    for (int t = 0; t < 4; t++)
        wrow[t] = WT + (size_t)(n0 + t * 16 + row) * D_MODEL + hi * 16;

    v8f acc[4];
    #pragma unroll
    for (int t = 0; t < 4; t++) acc[t] = (v8f){0.f, 0.f, 0.f, 0.f, 0.f, 0.f, 0.f, 0.f};

    #pragma unroll 2
    for (int k0 = 0; k0 < D_MODEL; k0 += 32) {
        BFrag a;
        const unsigned short* pa = xrow + k0 + hi * 16;
        a.q[0] = ld_g4(pa);                  // A elems 0..7  : K = k0+hi*16+i
        a.q[1] = ld_g4(pa + 8);              // A elems 8..15 : K = k0+hi*16+8+i
        #pragma unroll
        for (int t = 0; t < 4; t++) {
            BFrag b;
            b.q[0] = ld_g4(wrow[t] + k0);    // B elems: K = k0+hi*16 + 0..15
            b.q[1] = ld_g4(wrow[t] + k0 + 8);
            __builtin_prefetch(wrow[t] + k0 + 128, 0, 1);
            acc[t] = wmma_bf16(a.v, b.v, acc[t]);
        }
    }

    #pragma unroll
    for (int t = 0; t < 4; t++) {
        const int ng = n0 + t * 16 + row;     // output column
        const float bn = bias[ng];
        const int h = ng >> 6;
        const int d = ng & 63;
        if (mat < 2) {
            unsigned short* dst = (mat == 0) ? qh : kh;
            #pragma unroll
            for (int r = 0; r < 8; r++) {     // C: M = r + 8*hi, N = lane%16
                int m = m0 + 8 * hi + r;
                int bb = m >> 11, s = m & (SEQ - 1);
                dst[(((size_t)(bb * NUM_HEADS + h) * SEQ + s) << 6) + d] = bf16u(acc[t][r] + bn);
            }
        } else {
            int m  = m0 + 8 * hi;             // 8 consecutive s -> one 16B store
            int bb = m >> 11, s = m & (SEQ - 1);
            BFrag o;
            #pragma unroll
            for (int r = 0; r < 8; r++) o.u[r] = bf16u(acc[t][r] + bn);
            *reinterpret_cast<u32x4*>(vth + ((size_t)((bb * NUM_HEADS + h) * D_K + d) * SEQ + s)) = o.q[0];
        }
    }
}

// ---------------------------------------------------------------------------
// K2: attention. 1 block = (b, h, 32 query rows), 8 waves, 32x2048 f32 score
// panel resident in LDS (256KB of the WGP's 320KB).
// ---------------------------------------------------------------------------
__global__ void t5_attn(const unsigned short* __restrict__ qh,
                        const unsigned short* __restrict__ kh,
                        const unsigned short* __restrict__ vth,
                        const float* __restrict__ rel_table,
                        unsigned short* __restrict__ ctx,
                        float* __restrict__ attn_out,
                        float* __restrict__ pb_out) {
    extern __shared__ float lds[];
    float* scores   = lds;                       // 32*2048 f32
    float* bias_tab = lds + 32 * SEQ;            // 4095 (+1 pad)
    float* red      = bias_tab + 4096;           // 256
    float* rowred   = red + 256;                 // 32

    const int tid  = threadIdx.x;
    const int lane = tid & 31;
    const int wave = tid >> 5;
    const int row  = lane & 15;
    const int hi   = lane >> 4;

    const int b  = blockIdx.x / NUM_HEADS;
    const int h  = blockIdx.x % NUM_HEADS;
    const int q0 = blockIdx.y * 32;

    // phase 0: per-head bias value for every relative offset d in [-2047,2047]
    for (int i = tid; i < 2 * SEQ - 1; i += 256) {
        int d = i - (SEQ - 1);
        bias_tab[i] = rel_table[rel_bucket(d) * NUM_HEADS + h];
    }
    __syncthreads();

    const size_t bh = (size_t)(b * NUM_HEADS + h);
    const unsigned short* qbase = qh + bh * SEQ * D_K;
    const unsigned short* kbase = kh + bh * SEQ * D_K;

    // phase 1: scores = QK^T/8 + bias  (preload 4 A fragments, stream K tiles)
    BFrag afrag[2][2];
    #pragma unroll
    for (int rt = 0; rt < 2; rt++)
        #pragma unroll
        for (int ks = 0; ks < 2; ks++) {
            const unsigned short* p = qbase + (size_t)(q0 + rt * 16 + row) * D_K + ks * 32 + hi * 16;
            afrag[rt][ks].q[0] = ld_g4(p);
            afrag[rt][ks].q[1] = ld_g4(p + 8);
        }

    float* pbrow = pb_out + bh * SEQ * SEQ;
    for (int ct = wave; ct < SEQ / 16; ct += 8) {
        BFrag bfrag[2];
        const unsigned short* kp = kbase + (size_t)(ct * 16 + row) * D_K + hi * 16;
        bfrag[0].q[0] = ld_g4(kp);      bfrag[0].q[1] = ld_g4(kp + 8);
        bfrag[1].q[0] = ld_g4(kp + 32); bfrag[1].q[1] = ld_g4(kp + 40);
        const int kpos = ct * 16 + row;
        #pragma unroll
        for (int rt = 0; rt < 2; rt++) {
            v8f acc = {0.f, 0.f, 0.f, 0.f, 0.f, 0.f, 0.f, 0.f};
            acc = wmma_bf16(afrag[rt][0].v, bfrag[0].v, acc);
            acc = wmma_bf16(afrag[rt][1].v, bfrag[1].v, acc);
            #pragma unroll
            for (int r = 0; r < 8; r++) {
                int ml   = rt * 16 + r + 8 * hi;         // local row
                int qpos = q0 + ml;
                float bv = bias_tab[kpos - qpos + (SEQ - 1)];
                scores[ml * SEQ + kpos] = acc[r] * 0.125f + bv;
                pbrow[(size_t)qpos * SEQ + kpos] = bv;
            }
        }
    }
    __syncthreads();

    // phase 2: exact softmax over each of the 32 rows (8 threads per row)
    {
        const int r32 = tid >> 3;
        const int seg = tid & 7;
        float* srow = scores + r32 * SEQ + seg * 256;
        float mx = -3.0e38f;
        for (int i = 0; i < 256; i++) mx = fmaxf(mx, srow[i]);
        red[r32 * 8 + seg] = mx;
        __syncthreads();
        if (seg == 0) {
            float m = red[r32 * 8];
            #pragma unroll
            for (int i = 1; i < 8; i++) m = fmaxf(m, red[r32 * 8 + i]);
            rowred[r32] = m;
        }
        __syncthreads();
        float rmax = rowred[r32];
        float sum = 0.f;
        for (int i = 0; i < 256; i++) {
            float e = __expf(srow[i] - rmax);
            srow[i] = e;
            sum += e;
        }
        red[r32 * 8 + seg] = sum;
        __syncthreads();
        if (seg == 0) {
            float s = 0.f;
            #pragma unroll
            for (int i = 0; i < 8; i++) s += red[r32 * 8 + i];
            rowred[r32] = 1.0f / s;
        }
        __syncthreads();
        float inv = rowred[r32];
        float* arow = attn_out + (bh * SEQ + (size_t)(q0 + r32)) * SEQ + seg * 256;
        for (int i = 0; i < 256; i += 4) {
            float p0 = srow[i]     * inv;
            float p1 = srow[i + 1] * inv;
            float p2 = srow[i + 2] * inv;
            float p3 = srow[i + 3] * inv;
            srow[i] = p0; srow[i + 1] = p1; srow[i + 2] = p2; srow[i + 3] = p3;
            arow[i] = p0; arow[i + 1] = p1; arow[i + 2] = p2; arow[i + 3] = p3;
        }
    }
    __syncthreads();

    // phase 3: context = P @ V  (one 16x16 tile per wave: rt in {0,1}, nt in {0..3})
    {
        const int rt = wave >> 2;
        const int nt = wave & 3;
        const unsigned short* vbase = vth + (bh * D_K + (size_t)(nt * 16 + row)) * SEQ;
        const float* prow = scores + (rt * 16 + row) * SEQ;
        v8f acc = {0.f, 0.f, 0.f, 0.f, 0.f, 0.f, 0.f, 0.f};
        #pragma unroll 4
        for (int kb = 0; kb < SEQ; kb += 32) {
            BFrag a, bv;
            const float* pa = prow + kb + hi * 16;
            #pragma unroll
            for (int i = 0; i < 8; i++) a.v[i]     = (__bf16)pa[i];       // K = kb+hi*16+i
            #pragma unroll
            for (int i = 0; i < 8; i++) a.v[8 + i] = (__bf16)pa[8 + i];   // K = kb+hi*16+8+i
            const unsigned short* pv = vbase + kb + hi * 16;
            bv.q[0] = ld_g4(pv);
            bv.q[1] = ld_g4(pv + 8);
            acc = wmma_bf16(a.v, bv.v, acc);
        }
        const int dcol = h * D_K + nt * 16 + row;
        #pragma unroll
        for (int r = 0; r < 8; r++) {
            int s = q0 + rt * 16 + r + 8 * hi;
            ctx[((size_t)(b * SEQ + s)) * D_MODEL + dcol] = bf16u(acc[r]);
        }
    }
}

// ---------------------------------------------------------------------------
// K3: output projection: out = ctx @ Wo + bo (f32 out), 16x64 per wave
// ---------------------------------------------------------------------------
__global__ void t5_oproj(const unsigned short* __restrict__ ctx,
                         const unsigned short* __restrict__ wto,
                         const float* __restrict__ bo,
                         float* __restrict__ out) {
    const int lane = threadIdx.x & 31;
    const int m0 = blockIdx.x * 16;
    const int n0 = blockIdx.y * 64;
    const int row = lane & 15;
    const int hi  = lane >> 4;
    const unsigned short* xrow = ctx + (size_t)(m0 + row) * D_MODEL;
    const unsigned short* wrow[4];
    #pragma unroll
    for (int t = 0; t < 4; t++)
        wrow[t] = wto + (size_t)(n0 + t * 16 + row) * D_MODEL + hi * 16;

    v8f acc[4];
    #pragma unroll
    for (int t = 0; t < 4; t++) acc[t] = (v8f){0.f, 0.f, 0.f, 0.f, 0.f, 0.f, 0.f, 0.f};

    #pragma unroll 2
    for (int k0 = 0; k0 < D_MODEL; k0 += 32) {
        BFrag a;
        const unsigned short* pa = xrow + k0 + hi * 16;
        a.q[0] = ld_g4(pa);            // K = k0+hi*16   + 0..7
        a.q[1] = ld_g4(pa + 8);        // K = k0+hi*16+8 + 0..7
        #pragma unroll
        for (int t = 0; t < 4; t++) {
            BFrag b;
            b.q[0] = ld_g4(wrow[t] + k0);
            b.q[1] = ld_g4(wrow[t] + k0 + 8);
            __builtin_prefetch(wrow[t] + k0 + 128, 0, 1);
            acc[t] = wmma_bf16(a.v, b.v, acc[t]);
        }
    }
    #pragma unroll
    for (int t = 0; t < 4; t++) {
        const int ng = n0 + t * 16 + row;
        const float bn = bo[ng];
        #pragma unroll
        for (int r = 0; r < 8; r++)
            out[(size_t)(m0 + 8 * hi + r) * D_MODEL + ng] = acc[t][r] + bn;
    }
}

// ---------------------------------------------------------------------------
extern "C" void kernel_launch(void* const* d_in, const int* in_sizes, int n_in,
                              void* d_out, int out_size, void* d_ws, size_t ws_size,
                              hipStream_t stream) {
    const float* query = (const float*)d_in[0];
    const float* key_  = (const float*)d_in[1];
    const float* value = (const float*)d_in[2];
    const float* Wq = (const float*)d_in[3];
    const float* bq = (const float*)d_in[4];
    const float* Wk = (const float*)d_in[5];
    const float* bk = (const float*)d_in[6];
    const float* Wv = (const float*)d_in[7];
    const float* bv = (const float*)d_in[8];
    const float* Wo = (const float*)d_in[9];
    const float* bo = (const float*)d_in[10];
    const float* rel = (const float*)d_in[11];

    const size_t WBYTES = (size_t)D_MODEL * D_MODEL * sizeof(unsigned short);             // 2 MB
    const size_t XBYTES = (size_t)BATCH * SEQ * D_MODEL * sizeof(unsigned short);         // 8 MB
    const size_t HBYTES = (size_t)BATCH * NUM_HEADS * SEQ * D_K * sizeof(unsigned short); // 8 MB
    char* ws = (char*)d_ws;
    unsigned short* wtq = (unsigned short*)(ws);
    unsigned short* wtk = (unsigned short*)(ws + WBYTES);
    unsigned short* wtv = (unsigned short*)(ws + 2 * WBYTES);
    unsigned short* wto = (unsigned short*)(ws + 3 * WBYTES);
    char* p = ws + 4 * WBYTES;
    unsigned short* xq  = (unsigned short*)(p);             p += XBYTES;
    unsigned short* xk  = (unsigned short*)(p);             p += XBYTES;
    unsigned short* xv  = (unsigned short*)(p);             p += XBYTES;
    unsigned short* qh  = (unsigned short*)(p);             p += HBYTES;
    unsigned short* kh  = (unsigned short*)(p);             p += HBYTES;
    unsigned short* vth = (unsigned short*)(p);             p += HBYTES;
    unsigned short* ctx = (unsigned short*)(p);

    float* out  = (float*)d_out;
    float* attn = out + (size_t)BATCH * SEQ * D_MODEL;
    float* pb   = attn + (size_t)BATCH * NUM_HEADS * SEQ * SEQ;

    // K0a: weight transpose+convert (4 matrices)
    t5_wtrans<<<dim3(D_MODEL * D_MODEL / 256, 4), 256, 0, stream>>>(
        Wq, Wk, Wv, Wo, wtq, wtk, wtv, wto);

    // K0b: activation f32->bf16 (3 tensors, 4 elems/thread)
    t5_xcvt<<<dim3(BATCH * SEQ * D_MODEL / 1024, 3), 256, 0, stream>>>(
        query, key_, value, xq, xk, xv);

    // K1: QKV projections (one wave per 16x64 tile)
    t5_qkv<<<dim3(BATCH * SEQ / 16, D_MODEL / 64, 3), 32, 0, stream>>>(
        xq, xk, xv, wtq, wtk, wtv, bq, bk, bv, qh, kh, vth);

    // K2: attention (scores + bias + softmax + P@V), 256KB+ LDS per block
    size_t smem = (size_t)(32 * SEQ + 4096 + 256 + 32) * sizeof(float);
    t5_attn<<<dim3(BATCH * NUM_HEADS, SEQ / 32), 256, smem, stream>>>(
        qh, kh, vth, rel, ctx, attn, pb);

    // K3: output projection (one wave per 16x64 tile)
    t5_oproj<<<dim3(BATCH * SEQ / 16, D_MODEL / 64), 32, 0, stream>>>(
        ctx, wto, bo, out);
}
